// Decoder_24618752540743
// MI455X (gfx1250) — matrix operand order repaired and checked by reading, not compiled, not run on hardware
//
#include <hip/hip_runtime.h>

// MI455X (gfx1250) hierarchical ChebConv decoder.
//
// All node tiles are FEATURE-MAJOR (N, F, B=256): each node tile is one
// contiguous 16-32KB block (coalesced gathers / block-copy upsampling /
// contiguous async-LDS staging), and WMMA C/D fragments (fixed column, 8
// consecutive rows) are contiguous 32B runs in LDS -> ds_*_b128, no
// divergence, no runtime-stride address chains.
//
// Per Chebyshev order: fused kernel = CSR gather/recurrence (f32) + WMMA
// (v_wmma_f32_16x16x32_bf16, f32 accumulate); the accumulator tile is staged
// into LDS with global_load_async_to_lds_b128 (ASYNCcnt) overlapped with the
// gather (or pre-filled with bias on the k=0 step, so the WMMA loop has a
// single unconditional C path), and written back as coalesced b128 stores.

typedef __attribute__((ext_vector_type(16))) __bf16 v16bf;
typedef __attribute__((ext_vector_type(8)))  __bf16 v8bf;
typedef __attribute__((ext_vector_type(8)))  float  v8f;
typedef __attribute__((ext_vector_type(4)))  float  v4f;

#define A_STRIDE 40     // bf16 elems per A row (32 + 8 pad: 20-bank rotation)
#define C_STRIDE 264    // f32 elems per C column (256 + 8 pad)

// ---------------------------------------------------------------- FC ---------
// x = data(256,8) @ fc_w(8,704) + fc_b -> feature-major (22, 32, 256).
__global__ void fc_kernel(const float* __restrict__ data,
                          const float* __restrict__ fc_w,
                          const float* __restrict__ fc_b,
                          float* __restrict__ out) {
  int t = blockIdx.x * blockDim.x + threadIdx.x;   // 22*256 threads
  int n = t >> 8, b = t & 255;
  float d[8];
#pragma unroll
  for (int i = 0; i < 8; ++i) d[i] = data[b * 8 + i];
  for (int f = 0; f < 32; ++f) {
    int col = n * 32 + f;
    float s = fc_b[col];
#pragma unroll
    for (int i = 0; i < 8; ++i) s = fmaf(d[i], fc_w[i * 704 + col], s);
    out[(size_t)n * 8192 + f * 256 + b] = s;
  }
}

// ----------------------------------------------------------- upsample --------
// dst[n] = (relu?) src[up[n]] : contiguous block copy, float4 vectorized.
__global__ void upsample_kernel(float* __restrict__ dst,
                                const float* __restrict__ src,
                                const int* __restrict__ up, int SZ, int relu) {
  int n = blockIdx.x;
  const v4f* s = (const v4f*)(src + (size_t)up[n] * SZ);
  v4f* d = (v4f*)(dst + (size_t)n * SZ);
  int nq = SZ >> 2;
  for (int i = threadIdx.x; i < nq; i += blockDim.x) {
    v4f v = s[i];
    if (relu) {
#pragma unroll
      for (int c = 0; c < 4; ++c) v[c] = fmaxf(v[c], 0.f);
    }
    d[i] = v;
  }
}

// ------------------------------------------------------------- CSR build -----
__global__ void csr_zero(int* __restrict__ rowptr, int* __restrict__ degs, int N) {
  int i = blockIdx.x * blockDim.x + threadIdx.x;
  if (i <= N) rowptr[i] = 0;
  if (i < N) degs[i] = 0;
}
__global__ void csr_count(const int* __restrict__ src, const int* __restrict__ dst,
                          int E, int* __restrict__ rowptr, int* __restrict__ degs) {
  int e = blockIdx.x * blockDim.x + threadIdx.x;
  if (e < E) {
    atomicAdd(&degs[src[e]], 1);         // reference degree counts src side
    atomicAdd(&rowptr[dst[e] + 1], 1);
  }
}
__global__ void csr_scan(int* __restrict__ rowptr, int N) {
  if (blockIdx.x == 0 && threadIdx.x == 0) {
    int s = 0;
    for (int i = 0; i <= N; ++i) { s += rowptr[i]; rowptr[i] = s; }
  }
}
__global__ void csr_node(const int* __restrict__ degs, const int* __restrict__ rowptr,
                         float* __restrict__ dinv, int* __restrict__ cursor, int N) {
  int i = blockIdx.x * blockDim.x + threadIdx.x;
  if (i < N) {
    int d = degs[i];
    dinv[i] = (d > 0) ? rsqrtf((float)d) : 0.f;
    cursor[i] = rowptr[i];
  }
}
__global__ void csr_fill(const int* __restrict__ src, const int* __restrict__ dst,
                         int E, int* __restrict__ cursor, const float* __restrict__ dinv,
                         int* __restrict__ csr_src, float* __restrict__ csr_norm) {
  int e = blockIdx.x * blockDim.x + threadIdx.x;
  if (e < E) {
    int s = src[e], d = dst[e];
    int slot = atomicAdd(&cursor[d], 1);
    csr_src[slot]  = s;
    csr_norm[slot] = -(dinv[s] * dinv[d]);   // lambda_max=2 scaled Laplacian
  }
}

// ----------------------------------------------- fused Cheb step + WMMA ------
// One block (256 threads = 8 waves) per node; thread t owns batch row b=t.
//  mode 0: A = Tin[n]                      (Tx0)
//  mode 1: A = prop(Tin)[n]                (Tx1)
//  mode 2: A = 2*prop(Tin)[n] - Tprev[n]   (recurrence)
template <int F_IN>
__global__ void __launch_bounds__(256)
cheb_step(const float* __restrict__ Tin, const float* __restrict__ Tprev,
          float* __restrict__ Tout,
          const float* __restrict__ W, const float* __restrict__ bias,
          float* __restrict__ outbuf, int Fout, int FoutPad,
          const int* __restrict__ rowptr, const int* __restrict__ csr_src,
          const float* __restrict__ csr_norm, int mode, int init) {
  __shared__ alignas(16) __bf16 Atile[256 * A_STRIDE]; // A: row b x K32, padded
  __shared__ alignas(16) __bf16 Wt[32 * A_STRIDE];     // W^T: Wt[fo][ki], padded
  __shared__ alignas(16) float  Ctile[32 * C_STRIDE];  // accum, column-major

  const int t = threadIdx.x;
  const int n = blockIdx.x;
  const size_t SZ = (size_t)256 * F_IN;
  float* outn = outbuf + (size_t)n * 256 * FoutPad;    // (FoutPad, 256) f-major

  // ---- C-tile prologue (uniform branch, before the barrier) ---------------
  if (init) {
    // k==0: seed the accumulator with the bias (uniform scalar load/column).
    for (int i = t; i < 32 * 256; i += 256) {
      int fo = i >> 8;                                 // uniform per iteration
      float bv = (fo < Fout) ? bias[fo] : 0.f;
      Ctile[fo * C_STRIDE + (i & 255)] = bv;
    }
  } else {
    // async-stage the accumulator tile into LDS (overlaps with the gather);
    // each 1KB column is contiguous in both global and LDS.
    unsigned cbase = (unsigned)(size_t)(void*)&Ctile[0];   // LDS byte offset
    const char* g = (const char*)outn;
    int chunks = FoutPad * 64;                             // 16B chunks
    for (int c = t; c < chunks; c += 256) {
      int fo = c >> 6, ch = c & 63;
      unsigned lds = cbase + (unsigned)(fo * (C_STRIDE * 4) + ch * 16);
      const char* ga = g + fo * 1024 + ch * 16;
      asm volatile("global_load_async_to_lds_b128 %0, %1, off"
                   :: "v"(lds), "v"(ga) : "memory");
    }
  }

  // ---- stage W[k] transposed + zero-padded into LDS (bf16) ----------------
  for (int i = t; i < 32 * 32; i += 256) {
    int fo = i >> 5, ki = i & 31;
    float w = (ki < F_IN && fo < Fout) ? W[ki * Fout + fo] : 0.f;
    Wt[fo * A_STRIDE + ki] = (__bf16)w;
  }

  // ---- propagation / recurrence in f32; thread t owns (b=t, f=i) ----------
  float acc[F_IN];
  if (mode == 0) {
    const float* p = Tin + (size_t)n * SZ + t;
#pragma unroll
    for (int i = 0; i < F_IN; ++i) acc[i] = p[256 * i];
  } else {
#pragma unroll
    for (int i = 0; i < F_IN; ++i) acc[i] = 0.f;
    int beg = rowptr[n], end = rowptr[n + 1];
    for (int j = beg; j < end; ++j) {
      int s = csr_src[j];
      float w = csr_norm[j];
      const float* p = Tin + (size_t)s * SZ + t;
      if (j + 1 < end)   // hide next neighbor block's latency
        __builtin_prefetch(Tin + (size_t)csr_src[j + 1] * SZ + t, 0, 3);
#pragma unroll
      for (int i = 0; i < F_IN; ++i) acc[i] = fmaf(w, p[256 * i], acc[i]);
    }
    if (mode == 2) {
      const float* q = Tprev + (size_t)n * SZ + t;
#pragma unroll
      for (int i = 0; i < F_IN; ++i) acc[i] = 2.f * acc[i] - q[256 * i];
    }
    float* o = Tout + (size_t)n * SZ + t;
#pragma unroll
    for (int i = 0; i < F_IN; ++i) o[256 * i] = acc[i];   // Tx_k for next step
  }

  // ---- stage A row b=t as bf16 via b128 stores (zero-pad K if F_IN==16) ---
  {
    __bf16* arow = &Atile[t * A_STRIDE];
#pragma unroll
    for (int ch = 0; ch < F_IN / 8; ++ch) {
      v8bf pk;
#pragma unroll
      for (int e = 0; e < 8; ++e) pk[e] = (__bf16)acc[ch * 8 + e];
      *(v8bf*)&arow[ch * 8] = pk;
    }
    if (F_IN == 16) {
      v8bf z;
#pragma unroll
      for (int e = 0; e < 8; ++e) z[e] = (__bf16)0.f;
      *(v8bf*)&arow[16] = z;
      *(v8bf*)&arow[24] = z;
    }
  }

  if (!init) asm volatile("s_wait_asynccnt 0x0" ::: "memory");
  __syncthreads();

  // ---- WMMA: (256 x 32) @ (32 x Fout), 8 waves x 2 M-tiles ----------------
  const int lane = t & 31;
  const int wave = t >> 5;
  const int hi   = lane >> 4;      // lanes 16-31: other K-halves / M+8
  const int mrow = lane & 15;
  const int kbA  = hi * 8;         // A: K in [kbA,kbA+8) U [16+kbA,16+kbA+8)
  const int kbB  = hi * 8;         // B: K in [kbB,kbB+8) U [16+kbB,16+kbB+8)
  const int ntiles = (Fout + 15) >> 4;

#pragma unroll
  for (int mi = 0; mi < 2; ++mi) {
    int mt  = wave * 2 + mi;
    int row = mt * 16 + mrow;
    v8bf a0 = *(const v8bf*)&Atile[row * A_STRIDE + kbA];
    v8bf a1 = *(const v8bf*)&Atile[row * A_STRIDE + 16 + kbA];
    v16bf afrag;
#pragma unroll
    for (int e = 0; e < 8; ++e) { afrag[e] = a0[e]; afrag[8 + e] = a1[e]; }

    for (int nt = 0; nt < ntiles; ++nt) {
      int fo = nt * 16 + mrow;                       // this lane's column
      v8bf b0 = *(const v8bf*)&Wt[fo * A_STRIDE + kbB];
      v8bf b1 = *(const v8bf*)&Wt[fo * A_STRIDE + kbB + 16];
      v16bf bfrag;
#pragma unroll
      for (int e = 0; e < 8; ++e) { bfrag[e] = b0[e]; bfrag[8 + e] = b1[e]; }

      // C/D fragment: column fo, rows mt*16 + 8*hi + [0,8) -> contiguous LDS.
      const int cb = fo * C_STRIDE + mt * 16 + 8 * hi;
      v4f c0 = *(const v4f*)&Ctile[cb];
      v4f c1 = *(const v4f*)&Ctile[cb + 4];
      v8f c;
#pragma unroll
      for (int v = 0; v < 4; ++v) { c[v] = c0[v]; c[4 + v] = c1[v]; }

      c = __builtin_amdgcn_wmma_f32_16x16x32_bf16(
              false, afrag, false, bfrag, (short)0, c, false, false);

      // D back to LDS (per-wave rows disjoint -> no barrier needed here)
      v4f d0, d1;
#pragma unroll
      for (int v = 0; v < 4; ++v) { d0[v] = c[v]; d1[v] = c[4 + v]; }
      *(v4f*)&Ctile[cb]     = d0;
      *(v4f*)&Ctile[cb + 4] = d1;
    }
  }
  __syncthreads();

  // ---- coalesced b128 writeback of the accumulator tile -------------------
  v4f* gout = (v4f*)outn;
  int chunks = FoutPad * 64;                          // v4f chunks
  for (int c = t; c < chunks; c += 256) {
    int fo = c >> 6, ch = c & 63;
    gout[fo * 64 + ch] = *(const v4f*)&Ctile[fo * C_STRIDE + ch * 4];
  }
}

// ---------------------------------------------------- final transpose --------
// feature-major (5632, 4, 256) -> reference layout (256, 5632, 3).
// 8 nodes per block staged through LDS: coalesced b128 reads, contiguous
// 96B runs per thread on the write side.
__global__ void out_transpose(const float* __restrict__ ob, float* __restrict__ out) {
  __shared__ alignas(16) float tile[8 * 1024];
  int n0 = blockIdx.x * 8;
  const v4f* src = (const v4f*)(ob + (size_t)n0 * 1024);
  v4f* dst = (v4f*)tile;
  for (int i = threadIdx.x; i < 8 * 256; i += 256) dst[i] = src[i];
  __syncthreads();
  int b = threadIdx.x;
  float* d = out + ((size_t)b * 5632 + n0) * 3;
#pragma unroll
  for (int n = 0; n < 8; ++n) {
    d[n * 3 + 0] = tile[n * 1024 + 0 * 256 + b];
    d[n * 3 + 1] = tile[n * 1024 + 1 * 256 + b];
    d[n * 3 + 2] = tile[n * 1024 + 2 * 256 + b];
  }
}

// ------------------------------------------------------------- host side -----
static inline void run_level(int N, int E, const int* src, const int* dst,
                             const float* W, const float* bias,
                             int Fin, int Fout, int FoutPad,
                             float* T0, float* T1, float* T2, float* OB,
                             int* rowptr, int* degs, int* cursor, int* csrs,
                             float* dinv, float* csrn, hipStream_t stream) {
  int gz = (N + 1 + 255) / 256;
  int ge = (E + 255) / 256;
  int gn = (N + 255) / 256;
  csr_zero <<<gz, 256, 0, stream>>>(rowptr, degs, N);
  csr_count<<<ge, 256, 0, stream>>>(src, dst, E, rowptr, degs);
  csr_scan <<<1, 1, 0, stream>>>(rowptr, N);
  csr_node <<<gn, 256, 0, stream>>>(degs, rowptr, dinv, cursor, N);
  csr_fill <<<ge, 256, 0, stream>>>(src, dst, E, cursor, dinv, csrs, csrn);

  const int stride = Fin * Fout;
  auto launch = [&](int mode, int init, const float* Tin, const float* Tprev,
                    float* Tout, int k) {
    if (Fin == 32)
      cheb_step<32><<<N, 256, 0, stream>>>(Tin, Tprev, Tout, W + (size_t)k * stride,
                                           bias, OB, Fout, FoutPad,
                                           rowptr, csrs, csrn, mode, init);
    else
      cheb_step<16><<<N, 256, 0, stream>>>(Tin, Tprev, Tout, W + (size_t)k * stride,
                                           bias, OB, Fout, FoutPad,
                                           rowptr, csrs, csrn, mode, init);
  };
  launch(0, 1, T0, nullptr, nullptr, 0);       // out  = Tx0 @ W0 + bias
  launch(1, 0, T0, nullptr, T1, 1);            // Tx1 = L*Tx0 ; out += Tx1 @ W1
  float *a = T0, *b = T1, *c = T2;
  for (int k = 2; k < 6; ++k) {                // Tx2 = 2*L*Tx1 - Tx0
    launch(2, 0, b, a, c, k);
    float* tmp = a; a = b; b = c; c = tmp;
  }
}

extern "C" void kernel_launch(void* const* d_in, const int* in_sizes, int n_in,
                              void* d_out, int out_size, void* d_ws, size_t ws_size,
                              hipStream_t stream) {
  const float* data = (const float*)d_in[0];
  const int* src3 = (const int*)d_in[1];  const int* dst3 = (const int*)d_in[2];
  const int* src2 = (const int*)d_in[3];  const int* dst2 = (const int*)d_in[4];
  const int* src1 = (const int*)d_in[5];  const int* dst1 = (const int*)d_in[6];
  const int* src0 = (const int*)d_in[7];  const int* dst0 = (const int*)d_in[8];
  const int* up4  = (const int*)d_in[9];  const int* up3  = (const int*)d_in[10];
  const int* up2  = (const int*)d_in[11]; const int* up1  = (const int*)d_in[12];
  const float* fc_w = (const float*)d_in[13]; const float* fc_b = (const float*)d_in[14];
  const float* w1 = (const float*)d_in[15];  const float* b1 = (const float*)d_in[16];
  const float* w2 = (const float*)d_in[17];  const float* b2 = (const float*)d_in[18];
  const float* w3 = (const float*)d_in[19];  const float* b3 = (const float*)d_in[20];
  const float* w4 = (const float*)d_in[21];  const float* b4 = (const float*)d_in[22];

  // Workspace carve-up (floats). CT = 5632*256*16 (largest Tx state).
  const size_t CT = 23068672;
  const size_t OBSZ = 5767168;            // max of N*256*FoutPad over levels
  float* ws = (float*)d_ws;
  float* T0 = ws;
  float* T1 = ws + CT;
  float* T2 = ws + 2 * CT;
  float* OB = ws + 3 * CT;
  int* rowptr = (int*)(ws + 3 * CT + OBSZ);
  int* degs   = rowptr + 5633;
  int* cursor = degs + 5632;
  int* csrs   = cursor + 5632;
  float* dinv = (float*)(csrs + 33792);
  float* csrn = dinv + 5632;

  // FC -> (22,32,256) feature-major, then upsample to level-3 input (no relu).
  fc_kernel<<<22, 256, 0, stream>>>(data, fc_w, fc_b, OB);
  upsample_kernel<<<88, 256, 0, stream>>>(T0, OB, up4, 8192, 0);

  // Level 3: N=88, 32->32
  run_level(88, 528, src3, dst3, w1, b1, 32, 32, 32,
            T0, T1, T2, OB, rowptr, degs, cursor, csrs, dinv, csrn, stream);
  upsample_kernel<<<352, 256, 0, stream>>>(T0, OB, up3, 8192, 1);

  // Level 2: N=352, 32->16
  run_level(352, 2112, src2, dst2, w2, b2, 32, 16, 16,
            T0, T1, T2, OB, rowptr, degs, cursor, csrs, dinv, csrn, stream);
  upsample_kernel<<<1408, 256, 0, stream>>>(T0, OB, up2, 4096, 1);

  // Level 1: N=1408, 16->16
  run_level(1408, 8448, src1, dst1, w3, b3, 16, 16, 16,
            T0, T1, T2, OB, rowptr, degs, cursor, csrs, dinv, csrn, stream);
  upsample_kernel<<<5632, 256, 0, stream>>>(T0, OB, up1, 4096, 1);

  // Level 0: N=5632, 16->3 (pad 4), no relu; then transpose to (B,N,3).
  run_level(5632, 33792, src0, dst0, w4, b4, 16, 3, 4,
            T0, T1, T2, OB, rowptr, degs, cursor, csrs, dinv, csrn, stream);
  out_transpose<<<704, 256, 0, stream>>>(OB, (float*)d_out);
}
